// TnlAttention_12060268167672
// MI455X (gfx1250) — matrix operand order repaired and checked by reading, not compiled, not run on hardware
//
#include <hip/hip_runtime.h>
#include <hip/hip_bf16.h>

typedef __attribute__((ext_vector_type(16))) _Float16 v16h;
typedef __attribute__((ext_vector_type(8)))  _Float16 h8;
typedef __attribute__((ext_vector_type(8)))  float    v8f;
typedef __attribute__((ext_vector_type(4)))  float    f4;

#define BATCH 2
#define SEQ   2048
#define EMB   768
#define HEADS 12
#define HDIM  64
#define MDIM  4096   // BATCH*SEQ
#define KDIM  768
#define NDIM  768

// ---------------------------------------------------------------------------
// f32 -> f16 conversion of x (same layout)
// ---------------------------------------------------------------------------
__global__ void xcvt_kernel(const float* __restrict__ x, _Float16* __restrict__ xh)
{
    const int idx = blockIdx.x * 256 + threadIdx.x;       // MDIM*KDIM
    xh[idx] = (_Float16)x[idx];
}

// ---------------------------------------------------------------------------
// Weight convert + transpose: WT[n][k] = (f16) W[k][n]   (768x768)
// 32x32 tiles through LDS; block = 256 threads (32x8).
// ---------------------------------------------------------------------------
__global__ void wcvt_kernel(const float* __restrict__ W, _Float16* __restrict__ WT)
{
    __shared__ float tile[32][33];
    const int k0 = blockIdx.x * 32, n0 = blockIdx.y * 32;
    const int tx = threadIdx.x & 31, ty = threadIdx.x >> 5;
    #pragma unroll
    for (int r = 0; r < 32; r += 8)
        tile[ty + r][tx] = W[(size_t)(k0 + ty + r) * NDIM + n0 + tx];
    __syncthreads();
    #pragma unroll
    for (int r = 0; r < 32; r += 8)
        WT[(size_t)(n0 + ty + r) * KDIM + k0 + tx] = (_Float16)tile[tx][ty + r];
}

// ---------------------------------------------------------------------------
// WMMA GEMM over (4096 x 768) @ (768 x 768), templated:
//   MODE 0: A = xh (f16), silu, store f16 (B,H,N,D)         (Q, K)
//   MODE 1: A = xh (f16), silu, store f16 (B,H,D,N)         (V transposed)
//   MODE 2: A = AO*gate (f32->f16), no act, store f32 row-major (final @Wo)
// WT is f16, transposed (N x K) -> B fragments are contiguous 32B loads.
// Block = 128 threads (4 waves); each wave computes a 32x32 output tile.
// ---------------------------------------------------------------------------
template<int MODE>
__global__ void gemm_kernel(const _Float16* __restrict__ Xh,
                            const float* __restrict__ Xf,
                            const float* __restrict__ gate,
                            const _Float16* __restrict__ WT,
                            _Float16* __restrict__ outh,
                            float* __restrict__ outf)
{
    const int lane = threadIdx.x & 31;
    const int wave = threadIdx.x >> 5;
    const int m0 = blockIdx.x * 64 + (wave >> 1) * 32;
    const int n0 = blockIdx.y * 64 + (wave & 1) * 32;
    const int cl = lane & 15;
    const int hi = (lane & 16) ? 1 : 0;

    v8f acc[2][2] = {};

    for (int kb = 0; kb < KDIM; kb += 32) {
        v16h af[2], bf[2];
        #pragma unroll
        for (int mf = 0; mf < 2; ++mf) {
            const int row = m0 + mf * 16 + cl;
            if constexpr (MODE == 2) {
                const float* rp = Xf   + (size_t)row * KDIM + kb + hi * 16;
                const float* gp = gate + (size_t)row * KDIM + kb + hi * 16;
                f4 a0 = *(const f4*)(rp)      * *(const f4*)(gp);
                f4 a1 = *(const f4*)(rp + 4)  * *(const f4*)(gp + 4);
                f4 a2 = *(const f4*)(rp + 8) * *(const f4*)(gp + 8);
                f4 a3 = *(const f4*)(rp + 12) * *(const f4*)(gp + 12);
                #pragma unroll
                for (int i = 0; i < 4; ++i) {
                    af[mf][i]      = (_Float16)a0[i];
                    af[mf][4 + i]  = (_Float16)a1[i];
                    af[mf][8 + i]  = (_Float16)a2[i];
                    af[mf][12 + i] = (_Float16)a3[i];
                }
            } else {
                const _Float16* rp = Xh + (size_t)row * KDIM + kb + hi * 16;
                h8 a = *(const h8*)rp;
                h8 b = *(const h8*)(rp + 8);
                #pragma unroll
                for (int i = 0; i < 8; ++i) { af[mf][i] = a[i]; af[mf][8 + i] = b[i]; }
            }
        }
        #pragma unroll
        for (int nf = 0; nf < 2; ++nf) {
            const int col = n0 + nf * 16 + cl;
            bf[nf] = *(const v16h*)(WT + (size_t)col * KDIM + kb + hi * 16);
        }
        #pragma unroll
        for (int mf = 0; mf < 2; ++mf)
            #pragma unroll
            for (int nf = 0; nf < 2; ++nf)
                acc[mf][nf] = __builtin_amdgcn_wmma_f32_16x16x32_f16(
                    false, af[mf], false, bf[nf], (short)0, acc[mf][nf], false, false);
    }

    // Epilogue: C layout VGPR g -> M = g + 8*(lane>=16), N = lane%16
    #pragma unroll
    for (int mf = 0; mf < 2; ++mf) {
        #pragma unroll
        for (int nf = 0; nf < 2; ++nf) {
            #pragma unroll
            for (int g = 0; g < 8; ++g) {
                const int row = m0 + mf * 16 + g + hi * 8;
                const int col = n0 + nf * 16 + cl;
                float v = acc[mf][nf][g];
                if constexpr (MODE != 2) v = v / (1.0f + __expf(-v));   // silu
                if constexpr (MODE == 0) {
                    const int b = row >> 11, n = row & 2047;
                    const int h = col >> 6,  d = col & 63;
                    outh[(((size_t)(b * HEADS + h) * SEQ) + n) * HDIM + d] = (_Float16)v;
                } else if constexpr (MODE == 1) {
                    const int b = row >> 11, n = row & 2047;
                    const int h = col >> 6,  d = col & 63;
                    outh[(((size_t)(b * HEADS + h) * HDIM) + d) * SEQ + n] = (_Float16)v;
                } else {
                    outf[(size_t)row * NDIM + col] = v;
                }
            }
        }
    }
}

// ---------------------------------------------------------------------------
// Decay-masked attention: per (b,h):  O = (QK^T * decay) V
// Qh,Kh: f16 (BH, SEQ, 64);  Vth: f16 (BH, 64, SEQ).
// Block = 128 threads = 4 waves, each wave owns one 16-row block i0.
// Causality: only j-blocks with j0 <= i0 are visited.
// Decay factored: exp(s*(i-j)) = exp(s*(i-j0-31)) * exp(s*(31-(j-j0)))
//   -> 2 hoisted column factors + 8 row exps per j-step, branchless mask.
// ---------------------------------------------------------------------------
__global__ void attn_kernel(const _Float16* __restrict__ Qh, const _Float16* __restrict__ Kh,
                            const _Float16* __restrict__ Vth, const float* __restrict__ log_slope,
                            float* __restrict__ AO)
{
    __shared__ _Float16 sS[4][16 * 32];      // per-wave S staging (C-layout -> A-layout)
    const int lane = threadIdx.x & 31;
    const int wave = threadIdx.x >> 5;
    const int bh = blockIdx.x;               // 0..23
    const int b  = bh / HEADS, h = bh % HEADS;
    const int i0 = blockIdx.y * 64 + wave * 16;
    const int cl = lane & 15;
    const int hi = (lane & 16) ? 1 : 0;
    const float slope = log_slope[h];

    // column decay factors (constant over j: depend only on j - j0)
    const float fc0 = __expf(slope * (float)(31 - cl));        // nf = 0
    const float fc1 = __expf(slope * (float)(15 - cl));        // nf = 1

    // Q A-fragments for rows i0..i0+15 (K-dim = 64 -> two chunks of 32)
    const _Float16* qbase = Qh + ((size_t)bh * SEQ + i0 + cl) * HDIM + hi * 16;
    v16h qf[2];
    #pragma unroll
    for (int kc = 0; kc < 2; ++kc) {
        h8 a = *(const h8*)(qbase + kc * 32);
        h8 c = *(const h8*)(qbase + kc * 32 + 8);
        #pragma unroll
        for (int i = 0; i < 8; ++i) { qf[kc][i] = a[i]; qf[kc][8 + i] = c[i]; }
    }

    v8f o[4] = {};

    for (int j0 = 0; j0 <= i0; j0 += 32) {
        // S = Q_i * K_j^T  (16 x 32), f32 accumulate
        v8f s[2] = {};
        #pragma unroll
        for (int nf = 0; nf < 2; ++nf) {
            const _Float16* kcol = Kh + ((size_t)bh * SEQ + j0 + nf * 16 + cl) * HDIM + hi * 16;
            #pragma unroll
            for (int kc = 0; kc < 2; ++kc) {
                v16h kf = *(const v16h*)(kcol + kc * 32);
                s[nf] = __builtin_amdgcn_wmma_f32_16x16x32_f16(
                    false, qf[kc], false, kf, (short)0, s[nf], false, false);
            }
        }
        // row decay factors: fr[g] = exp(slope*(i0 + g + hi*8 - j0 - 31))
        const float basef = (float)(i0 + hi * 8 - j0 - 31);
        float fr[8];
        #pragma unroll
        for (int g = 0; g < 8; ++g)
            fr[g] = __expf(slope * (basef + (float)g));

        // decay + causal mask (branchless), stage to LDS as f16 row-major [16][32]
        #pragma unroll
        for (int nf = 0; nf < 2; ++nf) {
            const float fc = nf ? fc1 : fc0;
            #pragma unroll
            for (int g = 0; g < 8; ++g) {
                const int r   = g + hi * 8;
                const int row = i0 + r;
                const int col = j0 + nf * 16 + cl;
                float v = s[nf][g] * fr[g] * fc;
                v = (row >= col) ? v : 0.0f;                   // v_cndmask
                sS[wave][r * 32 + nf * 16 + cl] = (_Float16)v;
            }
        }
        // reload S in A-fragment layout (same wave: DS is in-order, no barrier)
        v16h sa;
        {
            const int ar = cl;
            const int ak = hi * 16;
            h8 lo  = *(const h8*)&sS[wave][ar * 32 + ak];
            h8 hi8 = *(const h8*)&sS[wave][ar * 32 + ak + 8];
            #pragma unroll
            for (int i = 0; i < 8; ++i) { sa[i] = lo[i]; sa[8 + i] = hi8[i]; }
        }
        // O += S * V_j   (V transposed f16: one contiguous 32B B-fragment load)
        #pragma unroll
        for (int nf = 0; nf < 4; ++nf) {
            v16h vf = *(const v16h*)(Vth + ((size_t)bh * HDIM + nf * 16 + cl) * SEQ + j0 + hi * 16);
            o[nf] = __builtin_amdgcn_wmma_f32_16x16x32_f16(
                false, sa, false, vf, (short)0, o[nf], false, false);
        }
    }

    // store O back to (B, N, E) layout
    #pragma unroll
    for (int nf = 0; nf < 4; ++nf) {
        #pragma unroll
        for (int g = 0; g < 8; ++g) {
            const int n = i0 + g + hi * 8;
            const int e = h * HDIM + nf * 16 + cl;
            AO[((size_t)b * SEQ + n) * EMB + e] = o[nf][g];
        }
    }
}

// ---------------------------------------------------------------------------
// Low-rank gate:  xg = x @ Wg1 (4096x16),  gate = sigmoid(xg @ Wg2) (4096x768)
// ---------------------------------------------------------------------------
__global__ void gate1_kernel(const float* __restrict__ x, const float* __restrict__ Wg1,
                             float* __restrict__ xg)
{
    const int idx = blockIdx.x * 256 + threadIdx.x;   // MDIM*16
    const int row = idx >> 4, c = idx & 15;
    float s = 0.0f;
    for (int k = 0; k < EMB; ++k)
        s += x[(size_t)row * EMB + k] * Wg1[(size_t)k * 16 + c];
    xg[idx] = s;
}

__global__ void gate2_kernel(const float* __restrict__ xg, const float* __restrict__ Wg2,
                             float* __restrict__ gate)
{
    const int row = blockIdx.y;                       // MDIM rows
    const int c   = blockIdx.x * 256 + threadIdx.x;   // 768 cols (3 blocks)
    float s = 0.0f;
    #pragma unroll
    for (int k = 0; k < 16; ++k)
        s += xg[row * 16 + k] * Wg2[(size_t)k * EMB + c];
    gate[(size_t)row * EMB + c] = 1.0f / (1.0f + __expf(-s));
}

// ---------------------------------------------------------------------------
// Post-LayerNorm over last dim (768), one block per row.
// ---------------------------------------------------------------------------
__global__ void ln_kernel(const float* __restrict__ Y, const float* __restrict__ w,
                          const float* __restrict__ bias, float* __restrict__ out)
{
    const int row = blockIdx.x;
    const int tid = threadIdx.x;
    const float* y = Y + (size_t)row * EMB;
    __shared__ float s1[256], s2[256];
    float a = 0.0f, q = 0.0f;
    for (int c = tid; c < EMB; c += 256) {
        const float v = y[c];
        a += v; q += v * v;
    }
    s1[tid] = a; s2[tid] = q;
    __syncthreads();
    for (int off = 128; off > 0; off >>= 1) {
        if (tid < off) { s1[tid] += s1[tid + off]; s2[tid] += s2[tid + off]; }
        __syncthreads();
    }
    const float mu  = s1[0] * (1.0f / EMB);
    const float var = s2[0] * (1.0f / EMB) - mu * mu;
    const float inv = rsqrtf(var + 1e-5f);
    for (int c = tid; c < EMB; c += 256)
        out[(size_t)row * EMB + c] = (y[c] - mu) * inv * w[c] + bias[c];
}

// ---------------------------------------------------------------------------
extern "C" void kernel_launch(void* const* d_in, const int* in_sizes, int n_in,
                              void* d_out, int out_size, void* d_ws, size_t ws_size,
                              hipStream_t stream)
{
    const float* x   = (const float*)d_in[0];
    const float* ls  = (const float*)d_in[1];
    const float* Wq  = (const float*)d_in[2];
    const float* Wk  = (const float*)d_in[3];
    const float* Wv  = (const float*)d_in[4];
    const float* Wo  = (const float*)d_in[5];
    const float* Wg1 = (const float*)d_in[6];
    const float* Wg2 = (const float*)d_in[7];
    const float* lw  = (const float*)d_in[8];
    const float* lb  = (const float*)d_in[9];
    float* out = (float*)d_out;

    char* w = (char*)d_ws;
    const size_t SZ   = (size_t)MDIM * EMB;     // 4096*768 elements
    const size_t WSZ  = (size_t)KDIM * NDIM;    // 768*768
    _Float16* xh  = (_Float16*)w;  w += SZ  * 2;
    _Float16* WqT = (_Float16*)w;  w += WSZ * 2;
    _Float16* WkT = (_Float16*)w;  w += WSZ * 2;
    _Float16* WvT = (_Float16*)w;  w += WSZ * 2;
    _Float16* WoT = (_Float16*)w;  w += WSZ * 2;
    _Float16* Qh  = (_Float16*)w;  w += SZ  * 2;
    _Float16* Kh  = (_Float16*)w;  w += SZ  * 2;
    _Float16* Vth = (_Float16*)w;  w += SZ  * 2;
    float* gt = (float*)w;         w += SZ * 4;
    float* AO = (float*)w;         w += SZ * 4;
    float* Yb = (float*)w;         w += SZ * 4;
    float* xg = (float*)w;         w += (size_t)MDIM * 16 * 4;

    dim3 ggrid(MDIM / 64, NDIM / 64);           // 64 x 12
    dim3 gblk(128);
    dim3 wgrid(KDIM / 32, NDIM / 32);           // 24 x 24

    xcvt_kernel<<<(MDIM * KDIM) / 256, 256, 0, stream>>>(x, xh);
    wcvt_kernel<<<wgrid, 256, 0, stream>>>(Wq, WqT);
    wcvt_kernel<<<wgrid, 256, 0, stream>>>(Wk, WkT);
    wcvt_kernel<<<wgrid, 256, 0, stream>>>(Wv, WvT);
    wcvt_kernel<<<wgrid, 256, 0, stream>>>(Wo, WoT);

    gemm_kernel<0><<<ggrid, gblk, 0, stream>>>(xh, nullptr, nullptr, WqT, Qh, nullptr);
    gemm_kernel<0><<<ggrid, gblk, 0, stream>>>(xh, nullptr, nullptr, WkT, Kh, nullptr);
    gemm_kernel<1><<<ggrid, gblk, 0, stream>>>(xh, nullptr, nullptr, WvT, Vth, nullptr);

    gate1_kernel<<<(MDIM * 16) / 256, 256, 0, stream>>>(x, Wg1, xg);
    gate2_kernel<<<dim3(EMB / 256, MDIM), 256, 0, stream>>>(xg, Wg2, gt);

    attn_kernel<<<dim3(BATCH * HEADS, SEQ / 64), 128, 0, stream>>>(Qh, Kh, Vth, ls, AO);

    gemm_kernel<2><<<ggrid, gblk, 0, stream>>>(nullptr, AO, gt, WoT, nullptr, Yb);

    ln_kernel<<<MDIM, 256, 0, stream>>>(Yb, lw, lb, out);
}
